// HGTPolicy_70403103916692
// MI455X (gfx1250) — compile-verified, matrix operand order checked by reading, not run on hardware
//
#include <hip/hip_runtime.h>
#include <hip/hip_bf16.h>
#include <math.h>

// ---------------------------------------------------------------------------
// Problem constants (fixed by the reference)
// ---------------------------------------------------------------------------
#define NR 100000
#define NO 100000
#define NT (NR + NO)
#define EC 200000          // edges per edge type
#define CDIM 128
#define HH 4
#define DD 32
#define BB 1024
#define OUTD 8

typedef __attribute__((ext_vector_type(16))) __bf16 v16bf;
typedef __attribute__((ext_vector_type(8)))  float  v8f;

__device__ __forceinline__ float gelu_exact(float x) {
    return 0.5f * x * (1.0f + erff(x * 0.70710678118654752440f));
}
__device__ __forceinline__ unsigned ordF(float f) {
    unsigned u = __float_as_uint(f);
    return (u & 0x80000000u) ? ~u : (u | 0x80000000u);
}
__device__ __forceinline__ float unordF(unsigned o) {
    return (o & 0x80000000u) ? __uint_as_float(o & 0x7FFFFFFFu) : __uint_as_float(~o);
}
// Wave-uniform predicate -> scalar branch (condition must be uniform).
__device__ __forceinline__ bool wave_uniform(bool c) {
    return __builtin_amdgcn_readfirstlane(c ? 1 : 0) != 0;
}

// Load one 16x32 bf16 A fragment (wave-striped per ISA 7.12.2) from a
// row-major fp32 matrix, optionally applying exact GELU.
template <int MODE>
__device__ __forceinline__ v16bf load_afrag(const float* __restrict__ xrow,
                                            int kOff, int kh) {
  v16bf a;
#pragma unroll
  for (int j = 0; j < 8; ++j) {
    const int kb = (j < 4) ? (kh * 8 + 2 * j) : (16 + kh * 8 + 2 * (j - 4));
    float f0 = xrow[kOff + kb];
    float f1 = xrow[kOff + kb + 1];
    if (MODE == 1) { f0 = gelu_exact(f0); f1 = gelu_exact(f1); }
    a[2 * j] = (__bf16)f0;
    a[2 * j + 1] = (__bf16)f1;
  }
  return a;
}

// ---------------------------------------------------------------------------
// Dense GEMM: Y[n,128] = epilogue( act(X)[n,128] @ W[128,128] + bias )
// MODE 0: Y = X@W + b                       (K/Q/V projection)
// MODE 1: Y = relu(g*(gelu(X)@W + b) + (1-g)*Xold),  g = sigmoid(*skipPtr)
// 256 threads = 8 waves; each wave owns TWO 16-row tiles (32 rows) so every
// B fragment read from LDS feeds two WMMAs. 256 rows per block.
// W staged in LDS as bf16, pre-transposed (WT[col][k]); stride 130 kills
// bank conflicts and makes (k,k+1) pairs single 32-bit DS reads.
// Epilogue: scalar (readfirstlane) full-tile guard; one base pointer per
// tile, all stores/loads at compile-time immediate offsets.
// ---------------------------------------------------------------------------
template <int MODE>
__global__ __launch_bounds__(256) void gemm128_kernel(
    const float* __restrict__ X, const float* __restrict__ W,
    const float* __restrict__ bias, float* __restrict__ Y, int nrows,
    const float* __restrict__ Xold, const float* __restrict__ skipPtr) {
  __shared__ __bf16 WT[128 * 130];
  const int tid = threadIdx.x;
#pragma unroll
  for (int i = 0; i < 64; ++i) {              // 16384 elems / 256 threads
    int idx = i * 256 + tid;                  // W is [k][c] row-major
    int k = idx >> 7, c = idx & 127;
    WT[c * 130 + k] = (__bf16)W[idx];
  }
  __syncthreads();

  const int lane = tid & 31;
  const int wave = tid >> 5;
  const int m    = lane & 15;
  const int kh   = (lane >> 4) & 1;
  const int rowBase = (blockIdx.x * 8 + wave) * 32;
  int rA0 = rowBase + m;        if (rA0 >= nrows) rA0 = nrows - 1;
  int rA1 = rowBase + 16 + m;   if (rA1 >= nrows) rA1 = nrows - 1;
  const float* x0 = X + (size_t)rA0 * CDIM;
  const float* x1 = X + (size_t)rA1 * CDIM;

  v8f acc0[8] = {};
  v8f acc1[8] = {};
#pragma unroll
  for (int kk = 0; kk < 4; ++kk) {
    const v16bf a0 = load_afrag<MODE>(x0, kk * 32, kh);
    const v16bf a1 = load_afrag<MODE>(x1, kk * 32, kh);
#pragma unroll
    for (int nt = 0; nt < 8; ++nt) {
      const int col = nt * 16 + m;
      v16bf b;
#pragma unroll
      for (int j = 0; j < 8; ++j) {
        const int kb = (j < 4) ? (kh * 8 + 2 * j) : (16 + kh * 8 + 2 * (j - 4));
        b[2 * j]     = WT[col * 130 + kk * 32 + kb];
        b[2 * j + 1] = WT[col * 130 + kk * 32 + kb + 1];
      }
      acc0[nt] = __builtin_amdgcn_wmma_f32_16x16x32_bf16(
          false, a0, false, b, (short)0, acc0[nt], false, false);
      acc1[nt] = __builtin_amdgcn_wmma_f32_16x16x32_bf16(
          false, a1, false, b, (short)0, acc1[nt], false, false);
    }
  }

  float gate = 0.0f;
  if (MODE == 1) gate = 1.0f / (1.0f + __expf(-skipPtr[0]));
  float bv[8];
#pragma unroll
  for (int nt = 0; nt < 8; ++nt) bv[nt] = bias[nt * 16 + m];

#pragma unroll
  for (int tile = 0; tile < 2; ++tile) {
    const v8f* acc = tile ? acc1 : acc0;
    const int base = rowBase + tile * 16;
    // per-lane base pointers; everything below is immediate offsets
    const size_t po = (size_t)(base + 8 * kh) * CDIM + m;
    float* yo = Y + po;
    const float* xo = (MODE == 1) ? (Xold + po) : nullptr;
    if (wave_uniform(base + 16 <= nrows)) {   // common case: scalar branch
#pragma unroll
      for (int r = 0; r < 8; ++r) {
#pragma unroll
        for (int nt = 0; nt < 8; ++nt) {
          float v = acc[nt][r] + bv[nt];
          if (MODE == 1) {
            const float old = xo[r * CDIM + nt * 16];
            v = gate * v + (1.0f - gate) * old;
            v = v > 0.0f ? v : 0.0f;          // post-layer ReLU
          }
          yo[r * CDIM + nt * 16] = v;
        }
      }
    } else {                                  // single ragged tile in the grid
#pragma unroll
      for (int r = 0; r < 8; ++r) {
        if (base + r + 8 * kh < nrows) {
#pragma unroll
          for (int nt = 0; nt < 8; ++nt) {
            float v = acc[nt][r] + bv[nt];
            if (MODE == 1) {
              const float old = xo[r * CDIM + nt * 16];
              v = gate * v + (1.0f - gate) * old;
              v = v > 0.0f ? v : 0.0f;
            }
            yo[r * CDIM + nt * 16] = v;
          }
        }
      }
    }
  }
}

// ---------------------------------------------------------------------------
// Per-head relation transform: Y[n, h*32+e] = sum_d X[n, h*32+d] * R[h][d][e]
// One K=32 bf16 WMMA covers a full head. Each wave owns 32 rows: every B
// fragment feeds two WMMAs. 4 heads x 2 column tiles x 2 row tiles per wave.
// ---------------------------------------------------------------------------
__global__ __launch_bounds__(256) void relmix_kernel(
    const float* __restrict__ X, const float* __restrict__ R,  // R: [4][32][32]
    float* __restrict__ Y, int nrows) {
  __shared__ __bf16 RT[4 * 32 * 33];
  const int tid = threadIdx.x;
#pragma unroll
  for (int i = 0; i < 16; ++i) {              // 4096 elems / 256 threads
    int idx = i * 256 + tid;
    int h = idx >> 10, rem = idx & 1023;
    int k = rem >> 5, c = rem & 31;
    RT[h * 1056 + c * 33 + k] = (__bf16)R[idx];
  }
  __syncthreads();

  const int lane = tid & 31;
  const int wave = tid >> 5;
  const int m    = lane & 15;
  const int kh   = (lane >> 4) & 1;
  const int rowBase = (blockIdx.x * 8 + wave) * 32;
  int rA0 = rowBase + m;        if (rA0 >= nrows) rA0 = nrows - 1;
  int rA1 = rowBase + 16 + m;   if (rA1 >= nrows) rA1 = nrows - 1;
  const float* x0 = X + (size_t)rA0 * CDIM;
  const float* x1 = X + (size_t)rA1 * CDIM;

  v8f acc0[8];
  v8f acc1[8];
#pragma unroll
  for (int h = 0; h < HH; ++h) {
    const v16bf a0 = load_afrag<0>(x0, h * 32, kh);
    const v16bf a1 = load_afrag<0>(x1, h * 32, kh);
#pragma unroll
    for (int nt = 0; nt < 2; ++nt) {
      const int col = nt * 16 + m;
      v16bf b;
#pragma unroll
      for (int j = 0; j < 8; ++j) {
        const int kb = (j < 4) ? (kh * 8 + 2 * j) : (16 + kh * 8 + 2 * (j - 4));
        b[2 * j]     = RT[h * 1056 + col * 33 + kb];
        b[2 * j + 1] = RT[h * 1056 + col * 33 + kb + 1];
      }
      v8f z = {};
      acc0[h * 2 + nt] = __builtin_amdgcn_wmma_f32_16x16x32_bf16(
          false, a0, false, b, (short)0, z, false, false);
      acc1[h * 2 + nt] = __builtin_amdgcn_wmma_f32_16x16x32_bf16(
          false, a1, false, b, (short)0, z, false, false);
    }
  }
#pragma unroll
  for (int tile = 0; tile < 2; ++tile) {
    const v8f* acc = tile ? acc1 : acc0;
    const int base = rowBase + tile * 16;
    float* yo = Y + (size_t)(base + 8 * kh) * CDIM + m;
    if (wave_uniform(base + 16 <= nrows)) {
#pragma unroll
      for (int r = 0; r < 8; ++r)
#pragma unroll
        for (int c = 0; c < 8; ++c)           // c = h*2+nt -> col h*32+nt*16
          yo[r * CDIM + (c >> 1) * 32 + (c & 1) * 16] = acc[c][r];
    } else {
#pragma unroll
      for (int r = 0; r < 8; ++r) {
        if (base + r + 8 * kh < nrows) {
#pragma unroll
          for (int c = 0; c < 8; ++c)
            yo[r * CDIM + (c >> 1) * 32 + (c & 1) * 16] = acc[c][r];
        }
      }
    }
  }
}

// ---------------------------------------------------------------------------
// Edge phase. One wave per edge; lane -> float4 chunk (head = lane>>3).
// ---------------------------------------------------------------------------
__global__ __launch_bounds__(256) void edge_logits_kernel(
    const float* __restrict__ Q, const float* __restrict__ Krel,
    const int* __restrict__ src, const int* __restrict__ dst, int nE,
    int dstOff, const float* __restrict__ prel, float* __restrict__ lg,
    unsigned* __restrict__ mxU) {
  const int w = blockIdx.x * 8 + (threadIdx.x >> 5);
  if (w >= nE) return;
  const int lane = threadIdx.x & 31;
  const int s = src[w], d = dst[w];
  const float4 qv = *(const float4*)(Q + (size_t)(d + dstOff) * CDIM + lane * 4);
  const float4 kv = *(const float4*)(Krel + (size_t)s * CDIM + lane * 4);
  float p = qv.x * kv.x + qv.y * kv.y + qv.z * kv.z + qv.w * kv.w;
  p += __shfl_xor(p, 1, 32);
  p += __shfl_xor(p, 2, 32);
  p += __shfl_xor(p, 4, 32);                  // sum over 8 lanes = one head
  if ((lane & 7) == 0) {
    const int h = lane >> 3;
    const float lv = p * prel[h] * 0.17677669529663687f;  // 1/sqrt(32)
    lg[(size_t)w * HH + h] = lv;
    atomicMax(&mxU[(size_t)(d + dstOff) * HH + h], ordF(lv));
  }
}

__global__ __launch_bounds__(256) void edge_exp_kernel(
    const int* __restrict__ dst, int nE, int dstOff,
    const float* __restrict__ lg, const unsigned* __restrict__ mxU,
    float* __restrict__ av, float* __restrict__ den) {
  const int idx = blockIdx.x * 256 + threadIdx.x;
  if (idx >= nE * HH) return;
  const int w = idx >> 2, h = idx & 3;
  const int g = (dst[w] + dstOff) * HH + h;
  const float m = unordF(mxU[g]);
  const float a = __expf(lg[idx] - m);
  av[idx] = a;
  atomicAdd(&den[g], a);
}

__global__ __launch_bounds__(256) void edge_agg_kernel(
    const float* __restrict__ Vrel, const int* __restrict__ src,
    const int* __restrict__ dst, int nE, int dstOff,
    const float* __restrict__ av, const float* __restrict__ den,
    float* __restrict__ agg) {
  const int w = blockIdx.x * 8 + (threadIdx.x >> 5);
  if (w >= nE) return;
  const int lane = threadIdx.x & 31;
  const int s = src[w], d = dst[w];
  const int h = lane >> 3;
  const float coef =
      av[(size_t)w * HH + h] / (den[(size_t)(d + dstOff) * HH + h] + 1e-16f);
  const float4 vv = *(const float4*)(Vrel + (size_t)s * CDIM + lane * 4);
  float* base = agg + (size_t)(d + dstOff) * CDIM + lane * 4;
  atomicAdd(base + 0, vv.x * coef);
  atomicAdd(base + 1, vv.y * coef);
  atomicAdd(base + 2, vv.z * coef);
  atomicAdd(base + 3, vv.w * coef);
}

// ---------------------------------------------------------------------------
// Pooling + head
// ---------------------------------------------------------------------------
__global__ __launch_bounds__(256) void pool_accum_kernel(
    const float* __restrict__ x, const int* __restrict__ batch_r,
    const int* __restrict__ batch_o, float* __restrict__ sums,
    float* __restrict__ cnt) {
  const int idx = blockIdx.x * 256 + threadIdx.x;
  if (idx >= NT * CDIM) return;
  const int node = idx >> 7, c = idx & 127;
  const int b = (node < NR) ? batch_r[node] : batch_o[node - NR];
  atomicAdd(&sums[b * CDIM + c], x[idx]);
  if (c == 0) atomicAdd(&cnt[b], 1.0f);
}

__global__ __launch_bounds__(128) void head_kernel(
    const float* __restrict__ sums, const float* __restrict__ cnt,
    const float* __restrict__ linW, const float* __restrict__ linb,
    float* __restrict__ out) {
  __shared__ float red[128];
  const int b = blockIdx.x, t = threadIdx.x;
  float c = cnt[b];
  c = c > 1.0f ? c : 1.0f;
  const float val = sums[b * CDIM + t] / c;
  for (int o = 0; o < OUTD; ++o) {
    red[t] = val * linW[t * OUTD + o];
    __syncthreads();
    for (int off = 64; off > 0; off >>= 1) {
      if (t < off) red[t] += red[t + off];
      __syncthreads();
    }
    if (t == 0) out[b * OUTD + o] = tanhf(red[0] + linb[o]);
    __syncthreads();
  }
}

// ---------------------------------------------------------------------------
// Host orchestration (graph-capture safe: only launches + async memcpy/memset)
// ---------------------------------------------------------------------------
extern "C" void kernel_launch(void* const* d_in, const int* in_sizes, int n_in,
                              void* d_out, int out_size, void* d_ws,
                              size_t ws_size, hipStream_t stream) {
  const float* x_robot = (const float*)d_in[0];
  const float* x_obj   = (const float*)d_in[1];
  const int*   ei[3]   = {(const int*)d_in[2], (const int*)d_in[3],
                          (const int*)d_in[4]};
  const int* batch_r = (const int*)d_in[5];
  const int* batch_o = (const int*)d_in[6];
  const float* kW = (const float*)d_in[7];
  const float* kb = (const float*)d_in[8];
  const float* qW = (const float*)d_in[9];
  const float* qb = (const float*)d_in[10];
  const float* vW = (const float*)d_in[11];
  const float* vb = (const float*)d_in[12];
  const float* aW = (const float*)d_in[13];
  const float* ab = (const float*)d_in[14];
  const float* skip = (const float*)d_in[15];
  const float* arel = (const float*)d_in[16];
  const float* mrel = (const float*)d_in[17];
  const float* prel = (const float*)d_in[18];
  const float* linW = (const float*)d_in[19];
  const float* linb = (const float*)d_in[20];

  // workspace layout (floats)
  float* ws = (float*)d_ws;
  size_t f = 0;
  float* xA = ws + f;  f += (size_t)NT * CDIM;
  float* xB = ws + f;  f += (size_t)NT * CDIM;
  float* Qb = ws + f;  f += (size_t)NT * CDIM;
  float* Tb = ws + f;  f += (size_t)NT * CDIM;          // K, then V, then agg
  float* Krel = ws + f; f += (size_t)(NR + 2 * NO) * CDIM;
  float* Vrel = ws + f; f += (size_t)(NR + 2 * NO) * CDIM;
  float* lg = ws + f;  f += (size_t)3 * EC * HH;
  float* av = ws + f;  f += (size_t)3 * EC * HH;
  unsigned* mxU = (unsigned*)(ws + f); f += (size_t)NT * HH;
  float* den = ws + f; f += (size_t)NT * HH;
  float* sums = ws + f; f += (size_t)BB * CDIM;
  float* cnt = ws + f;  f += (size_t)BB;

  // edge-type metadata (EDGE_META = [(0,1),(1,0),(1,1)])
  const int srcRowOff[3] = {0, NR, NR};       // row offset into K/V buffer
  const int dstOff[3]    = {NR, 0, NR};       // dst-type offset in node space
  const int nSrc[3]      = {NR, NO, NO};
  const size_t krelOff[3] = {0, (size_t)NR, (size_t)NR + NO};
  const int nPer[2] = {NR, NO};
  const int rowOff[2] = {0, NR};

  // x_all = concat(x_robot, x_obj)
  hipMemcpyAsync(xA, x_robot, (size_t)NR * CDIM * sizeof(float),
                 hipMemcpyDeviceToDevice, stream);
  hipMemcpyAsync(xA + (size_t)NR * CDIM, x_obj,
                 (size_t)NO * CDIM * sizeof(float), hipMemcpyDeviceToDevice,
                 stream);

  float* xin = xA;
  float* xout = xB;
  const int gemmGrid[2] = {(NR + 255) / 256, (NO + 255) / 256};
  const int edgeGrid = (EC + 7) / 8;
  const int expGrid = (EC * HH + 255) / 256;

  for (int l = 0; l < 2; ++l) {
    const size_t wOff = (size_t)l * 2;        // index into (2,2,...) params
    // ---- K projection, then per-edge-type relation transform of K
    for (int t = 0; t < 2; ++t)
      gemm128_kernel<0><<<gemmGrid[t], 256, 0, stream>>>(
          xin + (size_t)rowOff[t] * CDIM, kW + (wOff + t) * CDIM * CDIM,
          kb + (wOff + t) * CDIM, Tb + (size_t)rowOff[t] * CDIM, nPer[t],
          nullptr, nullptr);
    for (int e = 0; e < 3; ++e)
      relmix_kernel<<<(nSrc[e] + 255) / 256, 256, 0, stream>>>(
          Tb + (size_t)srcRowOff[e] * CDIM,
          arel + (size_t)(l * 3 + e) * HH * DD * DD, Krel + krelOff[e] * CDIM,
          nSrc[e]);
    // ---- V projection (reuses Tb), then relation transform of V
    for (int t = 0; t < 2; ++t)
      gemm128_kernel<0><<<gemmGrid[t], 256, 0, stream>>>(
          xin + (size_t)rowOff[t] * CDIM, vW + (wOff + t) * CDIM * CDIM,
          vb + (wOff + t) * CDIM, Tb + (size_t)rowOff[t] * CDIM, nPer[t],
          nullptr, nullptr);
    for (int e = 0; e < 3; ++e)
      relmix_kernel<<<(nSrc[e] + 255) / 256, 256, 0, stream>>>(
          Tb + (size_t)srcRowOff[e] * CDIM,
          mrel + (size_t)(l * 3 + e) * HH * DD * DD, Vrel + krelOff[e] * CDIM,
          nSrc[e]);
    // ---- Q projection
    for (int t = 0; t < 2; ++t)
      gemm128_kernel<0><<<gemmGrid[t], 256, 0, stream>>>(
          xin + (size_t)rowOff[t] * CDIM, qW + (wOff + t) * CDIM * CDIM,
          qb + (wOff + t) * CDIM, Qb + (size_t)rowOff[t] * CDIM, nPer[t],
          nullptr, nullptr);
    // ---- segment softmax over destinations (atomics)
    hipMemsetAsync(mxU, 0, (size_t)NT * HH * sizeof(unsigned), stream);
    hipMemsetAsync(den, 0, (size_t)NT * HH * sizeof(float), stream);
    for (int e = 0; e < 3; ++e)
      edge_logits_kernel<<<edgeGrid, 256, 0, stream>>>(
          Qb, Krel + krelOff[e] * CDIM, ei[e], ei[e] + EC, EC, dstOff[e],
          prel + (size_t)(l * 3 + e) * HH, lg + (size_t)e * EC * HH, mxU);
    for (int e = 0; e < 3; ++e)
      edge_exp_kernel<<<expGrid, 256, 0, stream>>>(
          ei[e] + EC, EC, dstOff[e], lg + (size_t)e * EC * HH, mxU,
          av + (size_t)e * EC * HH, den);
    // ---- weighted message aggregation into Tb (reused as agg)
    hipMemsetAsync(Tb, 0, (size_t)NT * CDIM * sizeof(float), stream);
    for (int e = 0; e < 3; ++e)
      edge_agg_kernel<<<edgeGrid, 256, 0, stream>>>(
          Vrel + krelOff[e] * CDIM, ei[e], ei[e] + EC, EC, dstOff[e],
          av + (size_t)e * EC * HH, den, Tb);
    // ---- output transform: gelu(agg)@aW + ab, sigmoid skip, ReLU
    for (int t = 0; t < 2; ++t)
      gemm128_kernel<1><<<gemmGrid[t], 256, 0, stream>>>(
          Tb + (size_t)rowOff[t] * CDIM, aW + (wOff + t) * CDIM * CDIM,
          ab + (wOff + t) * CDIM, xout + (size_t)rowOff[t] * CDIM, nPer[t],
          xin + (size_t)rowOff[t] * CDIM, skip + (wOff + t));
    float* tmp = xin; xin = xout; xout = tmp;
  }

  // ---- mean pooling by batch + linear head + tanh
  hipMemsetAsync(sums, 0, (size_t)BB * CDIM * sizeof(float), stream);
  hipMemsetAsync(cnt, 0, (size_t)BB * sizeof(float), stream);
  pool_accum_kernel<<<((size_t)NT * CDIM + 255) / 256, 256, 0, stream>>>(
      xin, batch_r, batch_o, sums, cnt);
  head_kernel<<<BB, 128, 0, stream>>>(sums, cnt, linW, linb, (float*)d_out);
}
